// LiquidHawkesModel_25220047962221
// MI455X (gfx1250) — matrix operand order repaired and checked by reading, not compiled, not run on hardware
//
#include <hip/hip_runtime.h>
#include <hip/hip_bf16.h>
#include <math.h>

// Model dims
#define B_   256
#define S_   512
#define IN_  64
#define H_   256
#define BB_  256
#define M_   64
#define R_   (B_ * S_)          // 131072 rows of (b,s)

typedef __attribute__((ext_vector_type(16))) _Float16 v16h;
typedef __attribute__((ext_vector_type(8)))  _Float16 v8h;
typedef __attribute__((ext_vector_type(8)))  float    v8f;

union V16U { v16h v; v8h h[2]; };

#define WMMA(a, b, c) \
  __builtin_amdgcn_wmma_f32_16x16x32_f16(false, (a), false, (b), (short)0, (c), false, false)

#define SCHED_FENCE() __builtin_amdgcn_sched_barrier(0)

// ---------- fragment loads (16-bit WMMA, contiguous k = 16*half + s, same in A and B) ----------
// A (16x32, row-major source, ld elements):
//   lane<16 : row=lane,     K = {0..15}
//   lane>=16: row=lane-16,  K = {16..31}
__device__ __forceinline__ v16h load_a_frag(const _Float16* base, int lda, int lane) {
  const int m = lane & 15, half = lane >> 4;
  const _Float16* p = base + m * lda + half * 16;
  V16U u;
  u.h[0] = *(const v8h*)(p);
  u.h[1] = *(const v8h*)(p + 8);
  return u.v;
}
// B (32x16) from W^T row-major (N rows x K cols, ld=K):
//   lane: col n = lane&15, K = 16*(lane>>4) + (0..15)  -> one contiguous 32B chunk
__device__ __forceinline__ v16h load_b_frag(const _Float16* base, int ldb, int lane) {
  const int n = lane & 15, half = lane >> 4;
  return *(const v16h*)(base + n * ldb + half * 16);
}

// Single-B-stream tile GEMM: preload ALL B fragments (global/L2 latency fully
// overlapped), double-buffer A fragments from LDS.
template <int NKT>
__device__ __forceinline__ v8f gemm_tile_preB(const _Float16* A, int lda,
                                              const _Float16* Bb, int ldb, int lane) {
  v16h b[NKT];
  #pragma unroll
  for (int kt = 0; kt < NKT; ++kt) b[kt] = load_b_frag(Bb + kt * 32, ldb, lane);
  v8f acc = {};
  v16h a = load_a_frag(A, lda, lane);
  #pragma unroll
  for (int kt = 0; kt < NKT - 1; ++kt) {
    v16h an = load_a_frag(A + (kt + 1) * 32, lda, lane);
    acc = WMMA(a, b[kt], acc);
    a = an;
  }
  return WMMA(a, b[NKT - 1], acc);
}

// 4-B-stream tile GEMM (shared A): depth-1 software pipeline — next iteration's
// loads are issued before the current WMMAs, so waits cover loads issued a full
// iteration earlier.
template <int NKT>
__device__ __forceinline__ void gemm_tile4_db(const _Float16* A, int lda,
                                              const _Float16* B0, const _Float16* B1,
                                              const _Float16* B2, const _Float16* B3,
                                              int ldb, int lane,
                                              v8f& c0, v8f& c1, v8f& c2, v8f& c3) {
  v16h a  = load_a_frag(A, lda, lane);
  v16h b0 = load_b_frag(B0, ldb, lane);
  v16h b1 = load_b_frag(B1, ldb, lane);
  v16h b2 = load_b_frag(B2, ldb, lane);
  v16h b3 = load_b_frag(B3, ldb, lane);
  #pragma unroll
  for (int kt = 0; kt < NKT - 1; ++kt) {
    v16h an  = load_a_frag(A + (kt + 1) * 32, lda, lane);
    v16h bn0 = load_b_frag(B0 + (kt + 1) * 32, ldb, lane);
    v16h bn1 = load_b_frag(B1 + (kt + 1) * 32, ldb, lane);
    v16h bn2 = load_b_frag(B2 + (kt + 1) * 32, ldb, lane);
    v16h bn3 = load_b_frag(B3 + (kt + 1) * 32, ldb, lane);
    c0 = WMMA(a, b0, c0);
    c1 = WMMA(a, b1, c1);
    c2 = WMMA(a, b2, c2);
    c3 = WMMA(a, b3, c3);
    a = an; b0 = bn0; b1 = bn1; b2 = bn2; b3 = bn3;
  }
  c0 = WMMA(a, b0, c0);
  c1 = WMMA(a, b1, c1);
  c2 = WMMA(a, b2, c2);
  c3 = WMMA(a, b3, c3);
}

// ---------- fast activations (v_rcp_f32 instead of IEEE divide) ----------
__device__ __forceinline__ float fast_rcp(float x)  { return __builtin_amdgcn_rcpf(x); }
__device__ __forceinline__ float fast_sig(float x)  { return fast_rcp(1.f + __expf(-x)); }
__device__ __forceinline__ float fast_tanh(float x) { return 1.f - 2.f * fast_rcp(1.f + __expf(2.f * x)); }
__device__ __forceinline__ float lecun_tanh(float x){ return 1.7159f * fast_tanh(0.666f * x); }
__device__ __forceinline__ float softplus_f(float x){ return (x > 20.f) ? x : __logf(1.f + __expf(x)); }
__device__ __forceinline__ float silu_f(float x)    { return x * fast_sig(x); }

// =====================================================================
// Prep: transpose f32 (K x N) -> f16 W^T (N x K) row-major
// =====================================================================
__global__ void transpose_to_f16(const float* __restrict__ src, _Float16* __restrict__ dst,
                                 int K, int N) {
  int idx = blockIdx.x * blockDim.x + threadIdx.x;
  if (idx >= K * N) return;
  int k = idx / N, n = idx - k * N;
  dst[n * K + k] = (_Float16)src[k * N + n];
}

// Pack ih_w0(64x16) | ah_w0(64x16) | ch_w0(64x8) | zeros -> Wh0^T (48 x 64) f16
// and biases -> hb0[48] f32
__global__ void pack_heads(const float* __restrict__ ihw0, const float* __restrict__ ahw0,
                           const float* __restrict__ chw0, const float* __restrict__ ihb0,
                           const float* __restrict__ ahb0, const float* __restrict__ chb0,
                           _Float16* __restrict__ wh0t, float* __restrict__ hb0) {
  int idx = blockIdx.x * blockDim.x + threadIdx.x;
  if (idx >= 48 * 64) return;
  int n = idx >> 6, k = idx & 63;
  float v = 0.f;
  if (n < 16)      v = ihw0[k * 16 + n];
  else if (n < 32) v = ahw0[k * 16 + (n - 16)];
  else if (n < 40) v = chw0[k * 8 + (n - 32)];
  wh0t[n * 64 + k] = (_Float16)v;
  if (k == 0) {
    float b = 0.f;
    if (n < 16)      b = ihb0[n];
    else if (n < 32) b = ahb0[n - 16];
    else if (n < 40) b = chb0[n - 32];
    hb0[n] = b;
  }
}

// =====================================================================
// Phase 1: CfC recurrent scan. 8 blocks x 256 threads (8 waves),
// each block owns 32 independent batch rows for all 512 steps.
// =====================================================================
#define ZLD  336   // zcat leading dim (320 + pad, mult of 8)
#define Z1LD 272   // z1/z2 leading dim (256 + pad, mult of 8)

__global__ __launch_bounds__(256)
void cfc_scan_kernel(const float* __restrict__ x, const float* __restrict__ tsp,
                     const float* __restrict__ hx,
                     const _Float16* __restrict__ W0t, const float* __restrict__ b0,
                     const _Float16* __restrict__ W1t, const float* __restrict__ b1,
                     const _Float16* __restrict__ ff1t, const float* __restrict__ ff1b,
                     const _Float16* __restrict__ ff2t, const float* __restrict__ ff2b,
                     const _Float16* __restrict__ tat,  const float* __restrict__ tab,
                     const _Float16* __restrict__ tbt,  const float* __restrict__ tbb,
                     _Float16* __restrict__ hseq, float* __restrict__ hn) {
  __shared__ __align__(32) _Float16 s_zcat[32 * ZLD];   // [x(64) | h(256)] per row, f16
  __shared__ __align__(32) _Float16 s_z1[32 * Z1LD];    // z1, then reused as h staging
  __shared__ __align__(32) _Float16 s_z2[32 * Z1LD];
  __shared__ float s_ts[32];

  const int tid  = threadIdx.x;
  const int lane = tid & 31;
  const int wave = tid >> 5;
  const int b0row = blockIdx.x * 32;
  const int nn   = lane & 15;
  const int half = lane >> 4;

  // init hidden-state region of zcat from hx (f32 -> f16)
  for (int i = tid; i < 32 * H_; i += 256) {
    int r = i >> 8, c = i & 255;
    s_zcat[r * ZLD + IN_ + c] = (_Float16)hx[(size_t)(b0row + r) * H_ + c];
  }
  __syncthreads();

  for (int t = 0; t < S_; ++t) {
    // ---- load x_t tile (32 x 64) and timespans ----
    {
      int r = tid >> 3, c = (tid & 7) * 8;
      const float* xp = x + (((size_t)(b0row + r)) * S_ + t) * IN_ + c;
      v8h hv;
      #pragma unroll
      for (int j = 0; j < 8; ++j) hv[j] = (_Float16)xp[j];
      *(v8h*)&s_zcat[r * ZLD + c] = hv;
    }
    if (tid < 32) s_ts[tid] = tsp[(size_t)(b0row + tid) * S_ + t];
    __syncthreads();

    // ---- GEMM0: z1 = lecun_tanh(zcat @ W0 + b0)   K=320 ----
    #pragma unroll 1
    for (int mt = 0; mt < 2; ++mt) {
      #pragma unroll 1
      for (int q = 0; q < 2; ++q) {
        int nt = wave * 2 + q;
        v8f acc = gemm_tile_preB<10>(&s_zcat[mt * 16 * ZLD], ZLD,
                                     W0t + nt * 16 * 320, 320, lane);
        float bias = b0[nt * 16 + nn];
        #pragma unroll
        for (int v = 0; v < 8; ++v) {
          float z = lecun_tanh(acc[v] + bias);
          s_z1[(mt * 16 + v + half * 8) * Z1LD + nt * 16 + nn] = (_Float16)z;
        }
        SCHED_FENCE();
      }
    }
    __syncthreads();

    // ---- GEMM1: z2 = lecun_tanh(z1 @ W1 + b1)   K=256 ----
    #pragma unroll 1
    for (int mt = 0; mt < 2; ++mt) {
      #pragma unroll 1
      for (int q = 0; q < 2; ++q) {
        int nt = wave * 2 + q;
        v8f acc = gemm_tile_preB<8>(&s_z1[mt * 16 * Z1LD], Z1LD,
                                    W1t + nt * 16 * 256, 256, lane);
        float bias = b1[nt * 16 + nn];
        #pragma unroll
        for (int v = 0; v < 8; ++v) {
          float z = lecun_tanh(acc[v] + bias);
          s_z2[(mt * 16 + v + half * 8) * Z1LD + nt * 16 + nn] = (_Float16)z;
        }
        SCHED_FENCE();
      }
    }
    __syncthreads();

    // ---- Heads: ff1/ff2/ta/tb (share A fragments), gate, stage h in s_z1 ----
    #pragma unroll 1
    for (int mt = 0; mt < 2; ++mt) {
      #pragma unroll 1
      for (int q = 0; q < 2; ++q) {
        int nt = wave * 2 + q;
        v8f cf1 = {}, cf2 = {}, cta = {}, ctb = {};
        gemm_tile4_db<8>(&s_z2[mt * 16 * Z1LD], Z1LD,
                         ff1t + nt * 16 * 256, ff2t + nt * 16 * 256,
                         tat  + nt * 16 * 256, tbt  + nt * 16 * 256,
                         256, lane, cf1, cf2, cta, ctb);
        int nb = nt * 16 + nn;
        float bf1 = ff1b[nb], bf2 = ff2b[nb], bta = tab[nb], btb = tbb[nb];
        #pragma unroll
        for (int v = 0; v < 8; ++v) {
          int mrow = mt * 16 + v + half * 8;
          float tsv = s_ts[mrow];
          float f1 = fast_tanh(cf1[v] + bf1);
          float f2 = fast_tanh(cf2[v] + bf2);
          float ti = fast_sig((cta[v] + bta) * tsv + (ctb[v] + btb));
          float h  = f1 + ti * (f2 - f1);
          s_zcat[mrow * ZLD + IN_ + nb] = (_Float16)h;   // feedback for next step
          s_z1[mrow * Z1LD + nb] = (_Float16)h;          // staging for coalesced store
          if (t == S_ - 1) hn[(size_t)(b0row + mrow) * H_ + nb] = h;
        }
        SCHED_FENCE();
      }
    }
    __syncthreads();

    // ---- coalesced h_seq store: 32 halves per thread, b128 stores ----
    {
      int r = tid >> 3, c0 = (tid & 7) * 32;
      _Float16* dst = &hseq[(((size_t)(b0row + r)) * S_ + t) * H_ + c0];
      const _Float16* srcp = &s_z1[r * Z1LD + c0];
      #pragma unroll
      for (int j = 0; j < 4; ++j)
        *(v8h*)(dst + j * 8) = *(const v8h*)(srcp + j * 8);
    }
    // next iteration's top-of-loop __syncthreads() protects s_z1/s_zcat reuse
    __syncthreads();
  }
}

// =====================================================================
// Phase 2: cfc = hseq @ proj_w + b  ->  packed head GEMM  ->  tiny MLPs
// 1024 blocks x 256 threads; each wave owns one 16-row tile of R.
// =====================================================================
#define CLD 72   // cfc LDS ld (64+8)
#define ALD 56   // silu LDS ld (48+8)

__global__ __launch_bounds__(256)
void head_kernel(const _Float16* __restrict__ hseq,
                 const _Float16* __restrict__ projt, const float* __restrict__ projb,
                 const _Float16* __restrict__ wh0t,  const float* __restrict__ hb0,
                 const float* __restrict__ ihw1, const float* __restrict__ ihb1,
                 const float* __restrict__ ahw1, const float* __restrict__ ahb1,
                 const float* __restrict__ chw1, const float* __restrict__ chb1,
                 float* __restrict__ out) {
  __shared__ __align__(32) _Float16 s_cfc[8][16 * CLD];
  __shared__ float s_act[8][16 * ALD];

  const int tid  = threadIdx.x;
  const int lane = tid & 31;
  const int wave = tid >> 5;
  const int nn   = lane & 15;
  const int half = lane >> 4;
  const size_t r0 = ((size_t)blockIdx.x * 8 + wave) * 16;

  // ---- proj GEMM: (16 x 256) @ (256 x 64), software-pipelined ----
  v8f c0 = {}, c1 = {}, c2 = {}, c3 = {};
  gemm_tile4_db<8>(hseq + r0 * H_, H_,
                   projt + 0 * 16 * H_, projt + 1 * 16 * H_,
                   projt + 2 * 16 * H_, projt + 3 * 16 * H_,
                   H_, lane, c0, c1, c2, c3);
  SCHED_FENCE();
  auto store_cfc = [&](v8f c, int nt) {
    float bias = projb[nt * 16 + nn];
    #pragma unroll
    for (int v = 0; v < 8; ++v)
      s_cfc[wave][(v + half * 8) * CLD + nt * 16 + nn] = (_Float16)(c[v] + bias);
  };
  store_cfc(c0, 0); store_cfc(c1, 1); store_cfc(c2, 2); store_cfc(c3, 3);
  // per-wave private LDS: same-wave DS ordering suffices (compiler inserts waits)

  // ---- packed head GEMM: (16 x 64) @ (64 x 48)  [ih|ah|ch|pad] + SiLU ----
  v8f a0 = {}, a1 = {}, a2 = {};
  #pragma unroll
  for (int kt = 0; kt < 2; ++kt) {
    v16h a = load_a_frag(&s_cfc[wave][kt * 32], CLD, lane);
    a0 = WMMA(a, load_b_frag(wh0t + 0 * 16 * 64 + kt * 32, 64, lane), a0);
    a1 = WMMA(a, load_b_frag(wh0t + 1 * 16 * 64 + kt * 32, 64, lane), a1);
    a2 = WMMA(a, load_b_frag(wh0t + 2 * 16 * 64 + kt * 32, 64, lane), a2);
  }
  SCHED_FENCE();
  auto store_act = [&](v8f c, int nt) {
    float bias = hb0[nt * 16 + nn];
    #pragma unroll
    for (int v = 0; v < 8; ++v) {
      float xin = c[v] + bias;
      s_act[wave][(v + half * 8) * ALD + nt * 16 + nn] = silu_f(xin);
    }
  };
  store_act(a0, 0); store_act(a1, 1); store_act(a2, 2);

  // ---- tiny output heads (VALU): lanes 0..15, one row each ----
  if (lane < 16) {
    const float* act = &s_act[wave][lane * ALD];
    size_t r = r0 + lane;
    #pragma unroll
    for (int j = 0; j < 3; ++j) {
      float acc = ihb1[j];
      #pragma unroll
      for (int i = 0; i < 16; ++i) acc += act[i] * ihw1[i * 3 + j];
      out[r * 3 + j] = softplus_f(acc);                       // intensities
    }
    float accA = ahb1[0];
    #pragma unroll
    for (int i = 0; i < 16; ++i) accA += act[16 + i] * ahw1[i];
    out[(size_t)R_ * 3 + r] = fast_tanh(accA);                // actions
    float accC = chb1[0];
    #pragma unroll
    for (int i = 0; i < 8; ++i) accC += act[32 + i] * chw1[i];
    out[(size_t)R_ * 4 + r] = fast_sig(accC);                 // confidence
  }
}

// =====================================================================
extern "C" void kernel_launch(void* const* d_in, const int* in_sizes, int n_in,
                              void* d_out, int out_size, void* d_ws, size_t ws_size,
                              hipStream_t stream) {
  const float* x    = (const float*)d_in[0];
  const float* tsp  = (const float*)d_in[1];
  const float* hx   = (const float*)d_in[2];
  const float* bbw0 = (const float*)d_in[3];  const float* bbb0 = (const float*)d_in[4];
  const float* bbw1 = (const float*)d_in[5];  const float* bbb1 = (const float*)d_in[6];
  const float* f1w  = (const float*)d_in[7];  const float* f1b  = (const float*)d_in[8];
  const float* f2w  = (const float*)d_in[9];  const float* f2b  = (const float*)d_in[10];
  const float* taw  = (const float*)d_in[11]; const float* tab_ = (const float*)d_in[12];
  const float* tbw  = (const float*)d_in[13]; const float* tbb_ = (const float*)d_in[14];
  const float* pjw  = (const float*)d_in[15]; const float* pjb  = (const float*)d_in[16];
  const float* ihw0 = (const float*)d_in[17]; const float* ihb0 = (const float*)d_in[18];
  const float* ihw1 = (const float*)d_in[19]; const float* ihb1 = (const float*)d_in[20];
  const float* ahw0 = (const float*)d_in[21]; const float* ahb0 = (const float*)d_in[22];
  const float* ahw1 = (const float*)d_in[23]; const float* ahb1 = (const float*)d_in[24];
  const float* chw0 = (const float*)d_in[25]; const float* chb0 = (const float*)d_in[26];
  const float* chw1 = (const float*)d_in[27]; const float* chb1 = (const float*)d_in[28];

  float* out = (float*)d_out;

  // workspace carve-out
  char* ws = (char*)d_ws;
  size_t off = 0;
  auto carve = [&](size_t bytes) -> void* {
    void* p = ws + off;
    off = (off + bytes + 255) & ~(size_t)255;
    return p;
  };
  _Float16* W0t   = (_Float16*)carve((size_t)256 * 320 * 2);
  _Float16* W1t   = (_Float16*)carve((size_t)256 * 256 * 2);
  _Float16* ff1t  = (_Float16*)carve((size_t)256 * 256 * 2);
  _Float16* ff2t  = (_Float16*)carve((size_t)256 * 256 * 2);
  _Float16* tat   = (_Float16*)carve((size_t)256 * 256 * 2);
  _Float16* tbt   = (_Float16*)carve((size_t)256 * 256 * 2);
  _Float16* projt = (_Float16*)carve((size_t)64 * 256 * 2);
  _Float16* wh0t  = (_Float16*)carve((size_t)48 * 64 * 2);
  float*    hb0   = (float*)carve((size_t)48 * 4);
  _Float16* hseq  = (_Float16*)carve((size_t)R_ * H_ * 2);

  // ---- prep: transpose/convert weights to f16 W^T ----
  struct { const float* s; _Float16* d; int K, N; } tr[7] = {
    { bbw0, W0t,  320, 256 }, { bbw1, W1t,  256, 256 },
    { f1w,  ff1t, 256, 256 }, { f2w,  ff2t, 256, 256 },
    { taw,  tat,  256, 256 }, { tbw,  tbt,  256, 256 },
    { pjw,  projt,256, 64  },
  };
  for (int i = 0; i < 7; ++i) {
    int n = tr[i].K * tr[i].N;
    transpose_to_f16<<<(n + 255) / 256, 256, 0, stream>>>(tr[i].s, tr[i].d, tr[i].K, tr[i].N);
  }
  pack_heads<<<(48 * 64 + 255) / 256, 256, 0, stream>>>(ihw0, ahw0, chw0, ihb0, ahb0, chb0,
                                                        wh0t, hb0);

  // ---- phase 1: recurrent scan (h_n written at offset 5*R) ----
  float* hn_out = out + (size_t)R_ * 5;
  cfc_scan_kernel<<<8, 256, 0, stream>>>(x, tsp, hx,
                                         W0t, bbb0, W1t, bbb1,
                                         ff1t, f1b, ff2t, f2b, tat, tab_, tbt, tbb_,
                                         hseq, hn_out);

  // ---- phase 2: projection + heads ----
  head_kernel<<<R_ / 128, 256, 0, stream>>>(hseq, projt, pjb, wh0t, hb0,
                                            ihw1, ihb1, ahw1, ahb1, chw1, chb1, out);
}
